// DOZSL_Random_23167053594875
// MI455X (gfx1250) — compile-verified, matrix operand order checked by reading, not run on hardware
//
#include <hip/hip_runtime.h>
#include <math.h>

// ---------------------------------------------------------------------------
// DOZSL scoring, fused for MI455X (gfx1250, wave32, WMMA f16).
//   out[b,n] = sigmoid(gamma - x2[b] - y2[n,rel[b]] + 2*dot(obj[b], x[n,rel[b],:]))
// Factor selection folded into zero-padded OBJ4[b,1024] -> scoring is one
// dense K=1024 GEMM. Encoder tile lives in LDS (never hits HBM).
// Rev 3: scheduler fences (__builtin_amdgcn_sched_barrier) so the double-
// buffered fragment loads actually stay a k-step ahead of the WMMAs, plus
// global_prefetch of the next Wt panel / OBJ4 window into WGP$ so residual
// just-in-time waits only pay cache-hit latency.
// ---------------------------------------------------------------------------

typedef _Float16 v16h __attribute__((ext_vector_type(16)));
typedef _Float16 v8h  __attribute__((ext_vector_type(8)));
typedef float    v8f  __attribute__((ext_vector_type(8)));

#define B_SZ  256
#define N_SZ  100000
#define D_SZ  256
#define KD    1024          // K * D
#define KF    4
#define TN    64            // n-rows per workgroup
#define XROW  1032          // padded LDS row stride in halfs (2064B -> 4-bank skew)

union H16 { v16h v; v8h h[2]; };

// ---------------------------------------------------------------------------
// Kernel 0: pca_w f32 [256][1024] -> f16 col-major Wt [1024][256] (L2 resident)
// ---------------------------------------------------------------------------
__global__ void k_wt(const float* __restrict__ w, _Float16* __restrict__ wt) {
  int r = blockIdx.x;                           // k index 0..255
  for (int c = threadIdx.x; c < KD; c += blockDim.x)
    wt[c * D_SZ + r] = (_Float16)w[r * KD + c];
}

// ---------------------------------------------------------------------------
// Kernel 1: per-b encoder slice -> OBJ4 (f16, factor-selected, zero padded)
//           and x2[b] = ||obj[b]||^2
// ---------------------------------------------------------------------------
__global__ __launch_bounds__(256) void k_obj(
    const int* __restrict__ sub, const int* __restrict__ rel,
    const float* __restrict__ E, const float* __restrict__ R,
    const float* __restrict__ w, const float* __restrict__ bias,
    _Float16* __restrict__ obj4, float* __restrict__ x2) {
  __shared__ float erow[D_SZ];
  __shared__ float red[256];
  const int b = blockIdx.x, d = threadIdx.x;
  const int sb = sub[b], kk = rel[b];
  erow[d] = E[(size_t)sb * D_SZ + d];
  __syncthreads();
  float acc = bias[kk * D_SZ + d];
  #pragma unroll 8
  for (int i = 0; i < D_SZ; ++i)
    acc = fmaf(erow[i], w[(size_t)i * KD + kk * D_SZ + d], acc);
  const float o = tanhf(acc) + R[(size_t)kk * D_SZ + d];
  #pragma unroll
  for (int k2 = 0; k2 < KF; ++k2)
    obj4[(size_t)b * KD + k2 * D_SZ + d] = (k2 == kk) ? (_Float16)o : (_Float16)0.0f;
  red[d] = o * o;
  __syncthreads();
  for (int s = 128; s > 0; s >>= 1) {
    if (d < s) red[d] += red[d + s];
    __syncthreads();
  }
  if (d == 0) x2[b] = red[0];
}

// ---------------------------------------------------------------------------
// Kernel 2: fused encoder GEMM + scoring GEMM + sigmoid epilogue.
// 256 threads = 8 waves. TN=64 n-rows per block.
// ---------------------------------------------------------------------------
__global__ __launch_bounds__(256, 1) void k_score(
    const float* __restrict__ E,         // [N][256] f32
    const _Float16* __restrict__ wt,     // [1024][256] f16 col-major
    const float* __restrict__ bias,      // [1024]
    const _Float16* __restrict__ obj4,   // [256][1024] f16
    const float* __restrict__ x2g,       // [256]
    const int* __restrict__ rel,         // [256]
    const float* __restrict__ gammap,    // [1]
    float* __restrict__ out) {           // [256][N]
  extern __shared__ char smem[];
  _Float16* Xt  = (_Float16*)smem;                        // [TN][XROW] f16
  float*    y2t = (float*)(smem + (size_t)TN * XROW * 2); // [TN][KF]
  float*    x2s = y2t + TN * KF;                          // [256]
  int*      rls = (int*)(x2s + B_SZ);                     // [256]

  const int tid  = threadIdx.x;
  const int lane = tid & 31;
  const int wave = tid >> 5;
  const int n0   = blockIdx.x * TN;
  const float gamma = gammap[0];

  const int lrow = lane & 15;           // fragment row/col within 16
  const int ksel = (lane >> 4) << 3;    // 0 (lanes 0-15) or 8 (lanes 16-31)

  if (tid < B_SZ) { x2s[tid] = x2g[tid]; rls[tid] = rel[tid]; }

  // ---------------- Stage A: X[TN][1024] = tanh(E @ W + b) -> LDS f16 -------
  // wave -> M-strip (16 rows) x 512-col half; E strip cached in registers.
  const int mstrip = wave >> 1;
  const int chalf  = (wave & 1) * 512;

  v16h ea[8];
  {
    const int  nrow = n0 + mstrip * 16 + lrow;
    const bool ok   = nrow < N_SZ;
    const float* er = E + (size_t)nrow * D_SZ;
    #pragma unroll
    for (int ks = 0; ks < 8; ++ks) {
      const int koff = ks * 32 + ksel;  // A layout: lanes<16 K 0-7/16-23, else 8-15/24-31
      v16h a;
      #pragma unroll
      for (int u = 0; u < 8; ++u) a[u]     = ok ? (_Float16)er[koff + u]      : (_Float16)0.0f;
      #pragma unroll
      for (int u = 0; u < 8; ++u) a[8 + u] = ok ? (_Float16)er[koff + 16 + u] : (_Float16)0.0f;
      ea[ks] = a;
    }
  }

  for (int g = 0; g < 8; ++g) {         // 4 col-tiles (64 cols) per pass
    const int cbase = chalf + g * 64;

    // Pull next g's 32KB Wt panel into WGP$ (waves 0/1 only, one per col-half).
    // global_prefetch has no dest VGPR / LOADcnt -> scheduler cannot sink it.
    if (wave < 2 && g < 7) {
      const char* nb = (const char*)(wt + (size_t)(cbase + 64) * D_SZ);
      #pragma unroll
      for (int p = 0; p < 8; ++p)
        __builtin_prefetch(nb + ((size_t)p * 32 + lane) * 128, 0, 3);
    }

    v8f acc[4] = {};
    H16 bf[2][4];
    #pragma unroll
    for (int t = 0; t < 4; ++t) {       // preload ks = 0
      const _Float16* wp = wt + (size_t)(cbase + t * 16 + lrow) * D_SZ + ksel;
      bf[0][t].h[0] = *(const v8h*)(wp);
      bf[0][t].h[1] = *(const v8h*)(wp + 16);
    }
    __builtin_amdgcn_sched_barrier(0);

    #pragma unroll
    for (int ks = 0; ks < 8; ++ks) {
      const int cur = ks & 1, nxt = cur ^ 1;
      if (ks < 7) {                     // issue next k-step's B loads first
        const int koff = (ks + 1) * 32 + ksel;
        #pragma unroll
        for (int t = 0; t < 4; ++t) {
          const _Float16* wp = wt + (size_t)(cbase + t * 16 + lrow) * D_SZ + koff;
          bf[nxt][t].h[0] = *(const v8h*)(wp);
          bf[nxt][t].h[1] = *(const v8h*)(wp + 16);
        }
      }
      __builtin_amdgcn_sched_barrier(0);  // keep loads above the WMMA batch
      #pragma unroll
      for (int t = 0; t < 4; ++t)
        acc[t] = __builtin_amdgcn_wmma_f32_16x16x32_f16(
            false, ea[ks], false, bf[cur][t].v, (short)0, acc[t], false, false);
      __builtin_amdgcn_sched_barrier(0);  // keep WMMAs above next iter's loads
    }

    // epilogue: bias + tanh, scatter to Xt (D layout: VGPR j -> rows j, j+8)
    const int rbase = mstrip * 16 + ksel;
    #pragma unroll
    for (int t = 0; t < 4; ++t) {
      const int col = cbase + t * 16 + lrow;
      const float bv = bias[col];
      #pragma unroll
      for (int j = 0; j < 8; ++j) {
        const float v = tanhf(acc[t][j] + bv);
        Xt[(size_t)(rbase + j) * XROW + col] = (_Float16)v;
      }
    }
  }
  __syncthreads();

  // ---------------- y2[n,k] = sum_d Xt[n][k*256+d]^2 ------------------------
  {
    const int nl = tid & 63, kf = tid >> 6;
    const _Float16* xp = Xt + (size_t)nl * XROW + kf * D_SZ;
    float s = 0.0f;
    #pragma unroll 4
    for (int j = 0; j < 32; ++j) {
      const int jr = (j + tid) & 31;    // rotate to spread LDS banks
      v8h hv = *(const v8h*)(xp + jr * 8);
      #pragma unroll
      for (int u = 0; u < 8; ++u) s += (float)hv[u] * (float)hv[u];
    }
    y2t[nl * KF + kf] = s;
  }
  __syncthreads();

  // ---------------- Stage B: S[256][TN] = OBJ4 @ Xt^T (K=1024) --------------
  // Double-buffered A (global OBJ4, L2-resident) and B (LDS Xt) fragments,
  // with sched fences + OBJ4 prefetch two k-chunks ahead.
  const int m0 = wave * 32;             // 2 M-tiles of b per wave
  v8f sacc[2][4] = {};
  H16 af[2][2], bf2[2][4];

  #pragma unroll
  for (int mt = 0; mt < 2; ++mt) {      // preload kc = 0
    const _Float16* ap = obj4 + (size_t)(m0 + mt * 16 + lrow) * KD + ksel;
    af[0][mt].h[0] = *(const v8h*)(ap);
    af[0][mt].h[1] = *(const v8h*)(ap + 16);
  }
  #pragma unroll
  for (int nt = 0; nt < 4; ++nt) {
    const _Float16* bp = Xt + (size_t)(nt * 16 + lrow) * XROW + ksel;
    bf2[0][nt].h[0] = *(const v8h*)(bp);
    bf2[0][nt].h[1] = *(const v8h*)(bp + 16);
  }
  __builtin_amdgcn_sched_barrier(0);

  #pragma unroll 2
  for (int kc = 0; kc < 32; ++kc) {
    const int cur = kc & 1, nxt = cur ^ 1;
    if (kc < 30) {                      // prefetch kc+2 window (1 line per b-row)
      const char* ap = (const char*)(obj4 + (size_t)(m0 + lane) * KD + (kc + 2) * 32);
      __builtin_prefetch(ap, 0, 3);
    }
    if (kc < 31) {                      // issue next k-chunk's fragment loads
      const int koff = (kc + 1) * 32 + ksel;
      #pragma unroll
      for (int mt = 0; mt < 2; ++mt) {
        const _Float16* ap = obj4 + (size_t)(m0 + mt * 16 + lrow) * KD + koff;
        af[nxt][mt].h[0] = *(const v8h*)(ap);
        af[nxt][mt].h[1] = *(const v8h*)(ap + 16);
      }
      #pragma unroll
      for (int nt = 0; nt < 4; ++nt) {
        const _Float16* bp = Xt + (size_t)(nt * 16 + lrow) * XROW + koff;
        bf2[nxt][nt].h[0] = *(const v8h*)(bp);
        bf2[nxt][nt].h[1] = *(const v8h*)(bp + 16);
      }
    }
    __builtin_amdgcn_sched_barrier(0);  // loads stay above this fence
    #pragma unroll
    for (int mt = 0; mt < 2; ++mt)
      #pragma unroll
      for (int nt = 0; nt < 4; ++nt)
        sacc[mt][nt] = __builtin_amdgcn_wmma_f32_16x16x32_f16(
            false, af[cur][mt].v, false, bf2[cur][nt].v, (short)0,
            sacc[mt][nt], false, false);
    __builtin_amdgcn_sched_barrier(0);  // WMMAs stay above next iter's loads
  }

  // ---------------- epilogue: gamma - x2 - y2 + 2*xy -> sigmoid -------------
  #pragma unroll
  for (int mt = 0; mt < 2; ++mt) {
    const int rb0 = m0 + mt * 16 + ksel;
    #pragma unroll
    for (int nt = 0; nt < 4; ++nt) {
      const int nl = nt * 16 + lrow;
      const int ng = n0 + nl;
      #pragma unroll
      for (int j = 0; j < 8; ++j) {
        const int b  = rb0 + j;
        const int kk = rls[b];
        const float sc = gamma - x2s[b] - y2t[nl * KF + kk] + 2.0f * sacc[mt][nt][j];
        const float sg = 1.0f / (1.0f + __expf(-sc));
        if (ng < N_SZ) out[(size_t)b * N_SZ + ng] = sg;
      }
    }
  }
}

// ---------------------------------------------------------------------------
// Host launcher. Inputs: sub, rel, init_embed, init_rel, pca_w, pca_b, gamma.
// Workspace: Wt f16 (512KB) | OBJ4 f16 (512KB) | x2 f32 (1KB)  ~= 1.05 MB.
// ---------------------------------------------------------------------------
extern "C" void kernel_launch(void* const* d_in, const int* in_sizes, int n_in,
                              void* d_out, int out_size, void* d_ws, size_t ws_size,
                              hipStream_t stream) {
  (void)in_sizes; (void)n_in; (void)out_size; (void)ws_size;
  const int*   sub   = (const int*)d_in[0];
  const int*   rel   = (const int*)d_in[1];
  const float* E     = (const float*)d_in[2];
  const float* R     = (const float*)d_in[3];
  const float* W     = (const float*)d_in[4];
  const float* bias  = (const float*)d_in[5];
  const float* gamma = (const float*)d_in[6];
  float*       out   = (float*)d_out;

  _Float16* wt   = (_Float16*)d_ws;
  _Float16* obj4 = (_Float16*)((char*)d_ws + (size_t)512 * 1024);
  float*    x2   = (float*)((char*)d_ws + (size_t)1024 * 1024);

  k_wt <<<D_SZ, 256, 0, stream>>>(W, wt);
  k_obj<<<B_SZ, 256, 0, stream>>>(sub, rel, E, R, W, bias, obj4, x2);

  const size_t smem = (size_t)TN * XROW * 2      // Xt
                    + (size_t)TN * KF * 4        // y2t
                    + (size_t)B_SZ * 4           // x2s
                    + (size_t)B_SZ * 4;          // rls
  const int nblocks = (N_SZ + TN - 1) / TN;
  k_score<<<nblocks, 256, smem, stream>>>(E, wt, bias, obj4, x2, rel, gamma, out);
}
